// WeldPointRegressionHGTNet_87153476370756
// MI455X (gfx1250) — matrix operand order repaired and checked by reading, not compiled, not run on hardware
//
#include <hip/hip_runtime.h>
#include <math.h>

typedef _Float16 f16;
typedef __attribute__((ext_vector_type(16))) _Float16 v16h;
typedef __attribute__((ext_vector_type(8)))  _Float16 v8h;
typedef __attribute__((ext_vector_type(8)))  float    v8f;

#define HIDC 64
#define NHEAD 8
#define HCH 64
#define HOUT 512
#define KMIX 4
#define NLAYER 3

__global__ void k_fill(float* p, long n, float v) {
  long i = (long)blockIdx.x * blockDim.x + threadIdx.x;
  if (i < n) p[i] = v;
}

// Transpose-convert weights: src [batch][Kdim][Ndim] f32 -> dst [batch][Ndim][Kdim] f16.
__global__ void k_convert_tr(const float* __restrict__ s, f16* __restrict__ d,
                             int Kdim, int Ndim, int batch) {
  long i = (long)blockIdx.x * blockDim.x + threadIdx.x;
  long per = (long)Kdim * Ndim;
  long total = per * batch;
  if (i >= total) return;
  long b = i / per;
  long rem = i - b * per;
  int n = (int)(rem / Kdim);
  int k = (int)(rem - (long)n * Kdim);
  d[i] = (f16)s[b * per + (size_t)k * Ndim + n];
}

__global__ void k_node_embed(const float* __restrict__ x, const float* __restrict__ W,
                             const float* __restrict__ b, f16* __restrict__ h16, int N) {
  int i = blockIdx.x * blockDim.x + threadIdx.x;
  if (i >= N * HIDC) return;
  int n = i >> 6, c = i & 63;
  const float* xr = x + (size_t)n * 7;
  float acc = b[c];
#pragma unroll
  for (int j = 0; j < 7; j++) acc += xr[j] * W[j * HIDC + c];
  h16[i] = (f16)fmaxf(acc, 0.f);
}

__global__ void k_edge_w(const float* __restrict__ x, const int* __restrict__ src,
                         const int* __restrict__ dst, const float* __restrict__ means,
                         const float* __restrict__ log_stds, float* __restrict__ wout, int E) {
  int e = blockIdx.x * blockDim.x + threadIdx.x;
  if (e >= E) return;
  int s = src[e], d = dst[e];
  float xs0 = x[(size_t)s * 7 + 0], xs1 = x[(size_t)s * 7 + 1];
  float xs2 = x[(size_t)s * 7 + 2], xs3 = x[(size_t)s * 7 + 3];
  float xd0 = x[(size_t)d * 7 + 0], xd1 = x[(size_t)d * 7 + 1];
  float xd2 = x[(size_t)d * 7 + 2], xd3 = x[(size_t)d * 7 + 3];
  float rx[4], ry[4];
  rx[0] = xd0 - xs0; ry[0] = xd1 - xs1;
  rx[1] = xd0 - xs2; ry[1] = xd1 - xs3;
  rx[2] = xd2 - xs0; ry[2] = xd3 - xs1;
  rx[3] = xd2 - xs2; ry[3] = xd3 - xs3;
  float u[8];
#pragma unroll
  for (int i = 0; i < 4; i++) {
    u[2 * i]     = sqrtf(rx[i] * rx[i] + ry[i] * ry[i]);
    u[2 * i + 1] = atan2f(ry[i], rx[i]);
  }
#pragma unroll
  for (int k = 0; k < KMIX; k++) {
    float ssum = 0.f;
#pragma unroll
    for (int j = 0; j < 8; j++) {
      float sd = expf(log_stds[k * 8 + j]) + 1e-6f;
      float z = (u[j] - means[k * 8 + j]) / sd;
      ssum += z * z;
    }
    wout[(size_t)e * KMIX + k] = expf(-0.5f * ssum);
  }
}

__global__ void k_edge_feat(const float* __restrict__ eattr, const float* __restrict__ w,
                            const float* __restrict__ eeW, const float* __restrict__ eeb,
                            const float* __restrict__ spW, const float* __restrict__ spb,
                            f16* __restrict__ ea16, int E) {
  int i = blockIdx.x * blockDim.x + threadIdx.x;
  if (i >= E * HIDC) return;
  int e = i >> 6, c = i & 63;
  float a1 = eeb[c], a2 = spb[c];
#pragma unroll
  for (int j = 0; j < 4; j++) {
    a1 += eattr[(size_t)e * 4 + j] * eeW[j * HIDC + c];
    a2 += w[(size_t)e * KMIX + j] * spW[j * HIDC + c];
  }
  ea16[i] = (f16)(fmaxf(a1, 0.f) + fmaxf(a2, 0.f));
}

// WMMA GEMM: one wave32 per 16x16 output tile, v_wmma_f32_16x16x32_f16 over K.
// A: [M][Kc] f16 row-major.  Bt: [Nc][Kc] f16 (transposed weights, N-major) so the
// B fragment is a contiguous 32B load per lane.
__global__ void k_gemm_f16(const f16* __restrict__ A, int lda,
                           const f16* __restrict__ Bt,
                           const float* __restrict__ bias,
                           f16* __restrict__ C16, float* __restrict__ C32, int ldc,
                           int M, int Nc, int Kc, int do_relu) {
  int lane = threadIdx.x & 31;
  int wave = threadIdx.x >> 5;
  int tilesN = Nc >> 4;
  int totalTiles = (M >> 4) * tilesN;
  int tile = blockIdx.x * (blockDim.x >> 5) + wave;
  if (tile >= totalTiles) return;
  int m0 = (tile / tilesN) << 4;
  int n0 = (tile % tilesN) << 4;
  int r  = lane & 15;
  int hi = lane >> 4;

  v8f acc = {};
  const f16* arow = A + (size_t)(m0 + r) * lda;
  const f16* bcol = Bt + (size_t)(n0 + r) * Kc + hi * 16;
  for (int k0 = 0; k0 < Kc; k0 += 32) {
    v8h alo = *(const v8h*)(arow + k0 + hi * 16);
    v8h ahi = *(const v8h*)(arow + k0 + hi * 16 + 8);
    v16h a;
#pragma unroll
    for (int i = 0; i < 8; i++) { a[i] = alo[i]; a[8 + i] = ahi[i]; }
    v16h b = *(const v16h*)(bcol + k0);
    acc = __builtin_amdgcn_wmma_f32_16x16x32_f16(false, a, false, b, (short)0, acc,
                                                 false, false);
  }
  float bv = bias ? bias[n0 + r] : 0.f;
#pragma unroll
  for (int q = 0; q < 8; q++) {
    float v = acc[q] + bv;
    if (do_relu) v = fmaxf(v, 0.f);
    size_t off = (size_t)(m0 + q + 8 * hi) * ldc + n0 + r;
    if (C16) C16[off] = (f16)v; else C32[off] = v;
  }
}

__device__ inline void atomicMaxFloat(float* addr, float val) {
  if (val >= 0.f) atomicMax((int*)addr, __float_as_int(val));
  else            atomicMin((unsigned int*)addr, (unsigned int)__float_as_int(val));
}

__global__ void k_alpha(const f16* __restrict__ Q16, const f16* __restrict__ K16,
                        const f16* __restrict__ Ep16, const int* __restrict__ src,
                        const int* __restrict__ dst, float* __restrict__ alpha,
                        float* __restrict__ m, int E) {
  int idx = blockIdx.x * blockDim.x + threadIdx.x;
  if (idx >= E * NHEAD) return;
  int e = idx >> 3, hd = idx & 7;
  int s = src[e], d = dst[e];
  const f16* qp = Q16 + (size_t)d * HOUT + hd * HCH;
  const f16* kp = K16 + (size_t)s * HOUT + hd * HCH;
  const f16* ep = Ep16 + (size_t)e * HOUT + hd * HCH;
  float acc = 0.f;
  for (int c = 0; c < HCH; c += 8) {
    v8h q  = *(const v8h*)(qp + c);
    v8h kk = *(const v8h*)(kp + c);
    v8h ee = *(const v8h*)(ep + c);
#pragma unroll
    for (int j = 0; j < 8; j++) acc += (float)q[j] * ((float)kk[j] + (float)ee[j]);
  }
  acc *= 0.125f;
  alpha[idx] = acc;
  atomicMaxFloat(&m[(size_t)d * NHEAD + hd], acc);
}

__global__ void k_ex(const int* __restrict__ dst, float* __restrict__ alpha,
                     const float* __restrict__ m, float* __restrict__ den, int E) {
  int idx = blockIdx.x * blockDim.x + threadIdx.x;
  if (idx >= E * NHEAD) return;
  int e = idx >> 3, hd = idx & 7;
  int d = dst[e];
  float ex = expf(alpha[idx] - m[(size_t)d * NHEAD + hd]);
  alpha[idx] = ex;
  atomicAdd(&den[(size_t)d * NHEAD + hd], ex);
}

__global__ void k_norm(const int* __restrict__ dst, float* __restrict__ alpha,
                       const float* __restrict__ den, int E) {
  int idx = blockIdx.x * blockDim.x + threadIdx.x;
  if (idx >= E * NHEAD) return;
  int e = idx >> 3, hd = idx & 7;
  alpha[idx] = alpha[idx] / (den[(size_t)dst[e] * NHEAD + hd] + 1e-16f);
}

__global__ void k_msg(const f16* __restrict__ V16, const f16* __restrict__ Ep16,
                      const int* __restrict__ src, const int* __restrict__ dst,
                      const float* __restrict__ alpha, float* __restrict__ outb, int E) {
  int idx = blockIdx.x * blockDim.x + threadIdx.x;
  if (idx >= E * HIDC) return;
  int e = idx >> 6, c = idx & 63;
  int s = src[e], d = dst[e];
  const f16* vp = V16 + (size_t)s * HOUT + c;
  const f16* ep = Ep16 + (size_t)e * HOUT + c;
  const float* ap = alpha + (size_t)e * NHEAD;
  float acc = 0.f;
#pragma unroll
  for (int hd = 0; hd < NHEAD; hd++)
    acc += ap[hd] * ((float)vp[hd * HCH] + (float)ep[hd * HCH]);
  atomicAdd(&outb[(size_t)d * HIDC + c], acc * 0.125f);
}

__global__ void k_combine(const float* __restrict__ outb, const float* __restrict__ S,
                          f16* __restrict__ h16, int N) {
  int i = blockIdx.x * blockDim.x + threadIdx.x;
  if (i >= N * HIDC) return;
  h16[i] = (f16)fmaxf(outb[i] + S[i], 0.f);
}

__global__ void k_ef(const f16* __restrict__ h16, const f16* __restrict__ ea16,
                     const int* __restrict__ src, const int* __restrict__ dst,
                     f16* __restrict__ ef16, int E) {
  int i = blockIdx.x * blockDim.x + threadIdx.x;
  if (i >= E * 192) return;
  int e = i / 192, c = i - e * 192;
  f16 v;
  if (c < 64)       v = h16[(size_t)src[e] * HIDC + c];
  else if (c < 128) v = h16[(size_t)dst[e] * HIDC + (c - 64)];
  else              v = ea16[(size_t)e * HIDC + (c - 128)];
  ef16[i] = v;
}

__global__ void k_pool_nodes(const f16* __restrict__ h16, const int* __restrict__ batch,
                             float* __restrict__ npool, float* __restrict__ cn, int N) {
  int i = blockIdx.x * blockDim.x + threadIdx.x;
  if (i >= N * HIDC) return;
  int n = i >> 6, c = i & 63;
  int b = batch[n];
  atomicAdd(&npool[(size_t)b * HIDC + c], (float)h16[i]);
  if (c == 0) atomicAdd(&cn[b], 1.0f);
}

__global__ void k_pool_edges(const f16* __restrict__ ea16, const int* __restrict__ src,
                             const int* __restrict__ batch, float* __restrict__ epool,
                             float* __restrict__ ce, int E) {
  int i = blockIdx.x * blockDim.x + threadIdx.x;
  if (i >= E * HIDC) return;
  int e = i >> 6, c = i & 63;
  int b = batch[src[e]];
  atomicAdd(&epool[(size_t)b * HIDC + c], (float)ea16[i]);
  if (c == 0) atomicAdd(&ce[b], 1.0f);
}

__global__ void k_head(const float* __restrict__ npool, const float* __restrict__ epool,
                       const float* __restrict__ cn, const float* __restrict__ ce,
                       const float* __restrict__ rW1, const float* __restrict__ rb1,
                       const float* __restrict__ rW2, const float* __restrict__ rb2,
                       const float* __restrict__ rW3, const float* __restrict__ rb3,
                       float* __restrict__ out, int G) {
  __shared__ float g[32 * 128];
  __shared__ float t1[32 * 64];
  __shared__ float t2[32 * 64];
  int tid = threadIdx.x;
  for (int i = tid; i < G * 64; i += blockDim.x) {
    int gr = i >> 6, c = i & 63;
    g[gr * 128 + c]      = npool[i] / fmaxf(cn[gr], 1.f);
    g[gr * 128 + 64 + c] = epool[i] / fmaxf(ce[gr], 1.f);
  }
  __syncthreads();
  for (int i = tid; i < G * 64; i += blockDim.x) {
    int gr = i >> 6, c = i & 63;
    float acc = rb1[c];
    for (int j = 0; j < 128; j++) acc += g[gr * 128 + j] * rW1[j * 64 + c];
    t1[i] = fmaxf(acc, 0.f);
  }
  __syncthreads();
  for (int i = tid; i < G * 64; i += blockDim.x) {
    int gr = i >> 6, c = i & 63;
    float acc = rb2[c];
    for (int j = 0; j < 64; j++) acc += t1[gr * 64 + j] * rW2[j * 64 + c];
    t2[i] = fmaxf(acc, 0.f);
  }
  __syncthreads();
  for (int i = tid; i < G * 2; i += blockDim.x) {
    int gr = i >> 1, o = i & 1;
    float acc = rb3[o];
    for (int j = 0; j < 64; j++) acc += t2[gr * 64 + j] * rW3[j * 2 + o];
    out[i] = tanhf(acc);
  }
}

extern "C" void kernel_launch(void* const* d_in, const int* in_sizes, int n_in,
                              void* d_out, int out_size, void* d_ws, size_t ws_size,
                              hipStream_t stream) {
  const float* x     = (const float*)d_in[0];
  const float* eattr = (const float*)d_in[1];
  const float* means = (const float*)d_in[2];
  const float* lstds = (const float*)d_in[3];
  const float* sp_W  = (const float*)d_in[4];
  const float* sp_b  = (const float*)d_in[5];
  const float* ne_W  = (const float*)d_in[6];
  const float* ne_b  = (const float*)d_in[7];
  const float* ee_W  = (const float*)d_in[8];
  const float* ee_b  = (const float*)d_in[9];
  const float* Wq    = (const float*)d_in[10];
  const float* bq    = (const float*)d_in[11];
  const float* Wk    = (const float*)d_in[12];
  const float* bk    = (const float*)d_in[13];
  const float* Wv    = (const float*)d_in[14];
  const float* bv    = (const float*)d_in[15];
  const float* We    = (const float*)d_in[16];
  const float* be    = (const float*)d_in[17];
  const float* Ws    = (const float*)d_in[18];
  const float* bs    = (const float*)d_in[19];
  const float* eu_W  = (const float*)d_in[20];
  const float* eu_b  = (const float*)d_in[21];
  const float* rW1   = (const float*)d_in[22];
  const float* rb1   = (const float*)d_in[23];
  const float* rW2   = (const float*)d_in[24];
  const float* rb2   = (const float*)d_in[25];
  const float* rW3   = (const float*)d_in[26];
  const float* rb3   = (const float*)d_in[27];
  const int*   ei    = (const int*)d_in[28];
  const int*   batch = (const int*)d_in[29];
  float* outp = (float*)d_out;

  const int N = in_sizes[0] / 7;
  const int E = in_sizes[1] / 4;
  const int G = out_size / 2;
  const int* src = ei;
  const int* dst = ei + E;

  char* base = (char*)d_ws;
  size_t off = 0;
  auto alloc = [&](size_t bytes) -> void* {
    void* p = base + off;
    off = (off + bytes + 255) & ~(size_t)255;
    return p;
  };
  f16*   h16   = (f16*)alloc((size_t)N * HIDC * 2);
  f16*   ea16  = (f16*)alloc((size_t)E * HIDC * 2);
  f16*   Q16   = (f16*)alloc((size_t)N * HOUT * 2);
  f16*   K16   = (f16*)alloc((size_t)N * HOUT * 2);
  f16*   V16   = (f16*)alloc((size_t)N * HOUT * 2);
  f16*   Ep16  = (f16*)alloc((size_t)E * HOUT * 2);
  f16*   ef16  = (f16*)alloc((size_t)E * 192 * 2);
  float* alpha = (float*)alloc((size_t)E * NHEAD * 4);
  float* mbuf  = (float*)alloc((size_t)N * NHEAD * 4);
  float* den   = (float*)alloc((size_t)N * NHEAD * 4);
  float* outb  = (float*)alloc((size_t)N * HIDC * 4);
  float* Sbuf  = (float*)alloc((size_t)N * HIDC * 4);
  float* wrbf  = (float*)alloc((size_t)E * KMIX * 4);
  f16*   WqT   = (f16*)alloc((size_t)NLAYER * HIDC * HOUT * 2);
  f16*   WkT   = (f16*)alloc((size_t)NLAYER * HIDC * HOUT * 2);
  f16*   WvT   = (f16*)alloc((size_t)NLAYER * HIDC * HOUT * 2);
  f16*   WeT   = (f16*)alloc((size_t)NLAYER * HIDC * HOUT * 2);
  f16*   WsT   = (f16*)alloc((size_t)NLAYER * HIDC * HIDC * 2);
  f16*   euT   = (f16*)alloc((size_t)NLAYER * 192 * HIDC * 2);
  float* npool = (float*)alloc((size_t)G * HIDC * 4);
  float* epool = (float*)alloc((size_t)G * HIDC * 4);
  float* cn    = (float*)alloc((size_t)G * 4);
  float* ce    = (float*)alloc((size_t)G * 4);

  const int BT = 256;
  auto cdiv = [](long a, long b) { return (int)((a + b - 1) / b); };

  { // weights -> transposed f16
    long n = (long)NLAYER * HIDC * HOUT;
    k_convert_tr<<<cdiv(n, BT), BT, 0, stream>>>(Wq, WqT, HIDC, HOUT, NLAYER);
    k_convert_tr<<<cdiv(n, BT), BT, 0, stream>>>(Wk, WkT, HIDC, HOUT, NLAYER);
    k_convert_tr<<<cdiv(n, BT), BT, 0, stream>>>(Wv, WvT, HIDC, HOUT, NLAYER);
    k_convert_tr<<<cdiv(n, BT), BT, 0, stream>>>(We, WeT, HIDC, HOUT, NLAYER);
    long ns = (long)NLAYER * HIDC * HIDC;
    k_convert_tr<<<cdiv(ns, BT), BT, 0, stream>>>(Ws, WsT, HIDC, HIDC, NLAYER);
    long nu = (long)NLAYER * 192 * HIDC;
    k_convert_tr<<<cdiv(nu, BT), BT, 0, stream>>>(eu_W, euT, 192, HIDC, NLAYER);
  }

  k_node_embed<<<cdiv((long)N * HIDC, BT), BT, 0, stream>>>(x, ne_W, ne_b, h16, N);
  k_edge_w<<<cdiv(E, BT), BT, 0, stream>>>(x, src, dst, means, lstds, wrbf, E);
  k_edge_feat<<<cdiv((long)E * HIDC, BT), BT, 0, stream>>>(eattr, wrbf, ee_W, ee_b,
                                                           sp_W, sp_b, ea16, E);

  for (int l = 0; l < NLAYER; l++) {
    int tilesQ = (N / 16) * (HOUT / 16);
    int tilesE = (E / 16) * (HOUT / 16);
    k_gemm_f16<<<cdiv(tilesQ, 8), BT, 0, stream>>>(h16, HIDC, WqT + (size_t)l * HIDC * HOUT,
        bq + (size_t)l * HOUT, Q16, nullptr, HOUT, N, HOUT, HIDC, 0);
    k_gemm_f16<<<cdiv(tilesQ, 8), BT, 0, stream>>>(h16, HIDC, WkT + (size_t)l * HIDC * HOUT,
        bk + (size_t)l * HOUT, K16, nullptr, HOUT, N, HOUT, HIDC, 0);
    k_gemm_f16<<<cdiv(tilesQ, 8), BT, 0, stream>>>(h16, HIDC, WvT + (size_t)l * HIDC * HOUT,
        bv + (size_t)l * HOUT, V16, nullptr, HOUT, N, HOUT, HIDC, 0);
    k_gemm_f16<<<cdiv(tilesE, 8), BT, 0, stream>>>(ea16, HIDC, WeT + (size_t)l * HIDC * HOUT,
        be + (size_t)l * HOUT, Ep16, nullptr, HOUT, E, HOUT, HIDC, 0);

    k_fill<<<cdiv((long)N * NHEAD, BT), BT, 0, stream>>>(mbuf, (long)N * NHEAD, -INFINITY);
    k_fill<<<cdiv((long)N * NHEAD, BT), BT, 0, stream>>>(den, (long)N * NHEAD, 0.f);
    k_fill<<<cdiv((long)N * HIDC, BT), BT, 0, stream>>>(outb, (long)N * HIDC, 0.f);

    k_alpha<<<cdiv((long)E * NHEAD, BT), BT, 0, stream>>>(Q16, K16, Ep16, src, dst, alpha, mbuf, E);
    k_ex<<<cdiv((long)E * NHEAD, BT), BT, 0, stream>>>(dst, alpha, mbuf, den, E);
    k_norm<<<cdiv((long)E * NHEAD, BT), BT, 0, stream>>>(dst, alpha, den, E);
    k_msg<<<cdiv((long)E * HIDC, BT), BT, 0, stream>>>(V16, Ep16, src, dst, alpha, outb, E);

    int tilesS = (N / 16) * (HIDC / 16);
    k_gemm_f16<<<cdiv(tilesS, 8), BT, 0, stream>>>(h16, HIDC, WsT + (size_t)l * HIDC * HIDC,
        bs + (size_t)l * HIDC, nullptr, Sbuf, HIDC, N, HIDC, HIDC, 0);
    k_combine<<<cdiv((long)N * HIDC, BT), BT, 0, stream>>>(outb, Sbuf, h16, N);

    k_ef<<<cdiv((long)E * 192, BT), BT, 0, stream>>>(h16, ea16, src, dst, ef16, E);
    int tilesU = (E / 16) * (HIDC / 16);
    k_gemm_f16<<<cdiv(tilesU, 8), BT, 0, stream>>>(ef16, 192, euT + (size_t)l * 192 * HIDC,
        eu_b + (size_t)l * HIDC, ea16, nullptr, HIDC, E, HIDC, 192, 1);
  }

  k_fill<<<cdiv((long)G * HIDC, BT), BT, 0, stream>>>(npool, (long)G * HIDC, 0.f);
  k_fill<<<cdiv((long)G * HIDC, BT), BT, 0, stream>>>(epool, (long)G * HIDC, 0.f);
  k_fill<<<1, BT, 0, stream>>>(cn, G, 0.f);
  k_fill<<<1, BT, 0, stream>>>(ce, G, 0.f);
  k_pool_nodes<<<cdiv((long)N * HIDC, BT), BT, 0, stream>>>(h16, batch, npool, cn, N);
  k_pool_edges<<<cdiv((long)E * HIDC, BT), BT, 0, stream>>>(ea16, src, batch, epool, ce, E);
  k_head<<<1, BT, 0, stream>>>(npool, epool, cn, ce, rW1, rb1, rW2, rb2, rW3, rb3, outp, G);
}
